// MultiHeadAttention_472446403334
// MI455X (gfx1250) — matrix operand order repaired and checked by reading, not compiled, not run on hardware
//
#include <hip/hip_runtime.h>
#include <hip/hip_bf16.h>

// ---------------------------------------------------------------------------
// MultiHeadAttention forward for MI455X (gfx1250, wave32, WMMA + TDM).
// B=2, S=2048, D_MODEL=1024, H=16, D_K=64.
// d_out = [ out (2,2048,1024) f32 | attn (2,16,2048,2048) f32 ]
// ---------------------------------------------------------------------------

#define D_MODEL 1024
#define SEQ     2048
#define BATCH   2
#define NH      16
#define DK      64
#define M_TOTAL (BATCH * SEQ)   // 4096 token rows
#define NEG_INF (-1e9f)

typedef __attribute__((ext_vector_type(16))) _Float16 v16h;
typedef __attribute__((ext_vector_type(8)))  float    v8f;
typedef unsigned int u32x4 __attribute__((ext_vector_type(4)));
typedef int          i32x8 __attribute__((ext_vector_type(8)));
typedef int          i32x4 __attribute__((ext_vector_type(4)));

// ---------------------------------------------------------------------------
// Fragment loaders for CDNA5 WMMA; A and B share contiguous k order (not ISA table order).
//
// A (16x32, f16): lane(0-15)=row M; lane half selects K block:
//                 K = lhi*16 + {0..15}   (lhi = lane>=16)
// B (32x16, f16): lane(0-15)=col N; lane half selects K block:
//                 K = lhi*16 + {0..15}
// C/D (16x16 f32, 8 VGPRs): element r -> (M = r + 8*lhi, N = lane&15)
// ---------------------------------------------------------------------------

// A fragment from row-major f32 source: element (m,k) = base[m*ld + k]
__device__ inline v16h frag_A_f32(const float* __restrict__ base, int ld,
                                  int m0, int k0, int lane) {
  const int lm = lane & 15, lhi = lane >> 4;
  const float* row = base + (size_t)(m0 + lm) * ld + k0;
  v16h a;
#pragma unroll
  for (int h = 0; h < 16; ++h) a[h] = (_Float16)row[lhi * 16 + h];
  return a;
}

// A fragment from row-major f16 source
__device__ inline v16h frag_A_f16(const _Float16* __restrict__ base, int ld,
                                  int m0, int k0, int lane) {
  const int lm = lane & 15, lhi = lane >> 4;
  const _Float16* row = base + (size_t)(m0 + lm) * ld + k0;
  v16h a;
#pragma unroll
  for (int h = 0; h < 16; ++h) a[h] = row[lhi * 16 + h];
  return a;
}

// A fragment from (generic/LDS) f32 source
__device__ inline v16h frag_A_lds(const float* base, int ld,
                                  int m0, int k0, int lane) {
  const int lm = lane & 15, lhi = lane >> 4;
  const float* row = base + (size_t)(m0 + lm) * ld + k0;
  v16h a;
#pragma unroll
  for (int h = 0; h < 16; ++h) a[h] = (_Float16)row[lhi * 16 + h];
  return a;
}

// B fragment where B(k,n) = src[n*ld + k]  (src stored N-major: W, or K for QK^T)
__device__ inline v16h frag_B_nk_f32(const float* __restrict__ base, int ld,
                                     int n0, int k0, int lane) {
  const int ln = lane & 15, lhi = lane >> 4;
  const float* p = base + (size_t)(n0 + ln) * ld + k0 + lhi * 16;
  v16h b;
#pragma unroll
  for (int h = 0; h < 16; ++h) b[h] = (_Float16)p[h];
  return b;
}

__device__ inline v16h frag_B_nk_f16(const _Float16* __restrict__ base, int ld,
                                     int n0, int k0, int lane) {
  const int ln = lane & 15, lhi = lane >> 4;
  const _Float16* p = base + (size_t)(n0 + ln) * ld + k0 + lhi * 16;
  v16h b;
#pragma unroll
  for (int h = 0; h < 16; ++h) b[h] = p[h];
  return b;
}

// B fragment where B(k,n) = src[k*ld + n] (row-major KxN), generic/LDS f16
__device__ inline v16h frag_B_kn_lds(const _Float16* base, int ld,
                                     int k0, int n0, int lane) {
  const int ln = lane & 15, lhi = lane >> 4;
  const _Float16* p = base + (size_t)(k0 + lhi * 16) * ld + n0 + ln;
  v16h b;
#pragma unroll
  for (int h = 0; h < 16; ++h) b[h] = p[(size_t)h * ld];
  return b;
}

// ---------------------------------------------------------------------------
// Tensor Data Mover: DMA a 2-D f16 tile (rows x cols) from global memory
// (row stride = row_stride elements) into contiguous LDS at lds_off.
// D# built per CDNA5 ISA 8.3/8.4: group0 {count=1, lds_addr, global_addr,
// type=2}; group1 {data_size=2B, tensor dims, tile dims, dim0 stride}.
// This toolchain's builtin takes 6 args (g0, g1, g2, g3, g_extra, cpol).
// ---------------------------------------------------------------------------
__device__ inline void tdm_load_2d_f16(unsigned lds_off, const void* gptr,
                                       int rows, int cols, int row_stride) {
  const unsigned long long ga = (unsigned long long)(uintptr_t)gptr;
  u32x4 g0;
  g0[0] = 1u;                                              // count=1, user D#
  g0[1] = lds_off;                                         // lds_addr (bytes)
  g0[2] = (unsigned)(ga & 0xffffffffu);                    // global_addr lo
  g0[3] = (unsigned)((ga >> 32) & 0x1ffffffu) | (2u << 30);// addr hi | type=2
  i32x8 g1;
  g1[0] = (1 << 16);                                       // data_size = 2 B
  g1[1] = (cols & 0xffff) << 16;                           // tensor_dim0 lo16
  g1[2] = ((cols >> 16) & 0xffff) | ((rows & 0xffff) << 16); // d0 hi | d1 lo
  g1[3] = ((rows >> 16) & 0xffff) | ((cols & 0xffff) << 16); // d1 hi | tile_dim0
  g1[4] = (rows & 0xffff);                                 // tile_dim1 (dim2=0)
  g1[5] = row_stride;                                      // dim0_stride lo32
  g1[6] = 0;                                               // stride hi bits
  g1[7] = 0;
  const i32x4 z4 = {0, 0, 0, 0};
  const i32x8 z8 = {0, 0, 0, 0, 0, 0, 0, 0};
  __builtin_amdgcn_tensor_load_to_lds(g0, g1, z4, z4, z8, 0);
}

// ---------------------------------------------------------------------------
// Kernel 1: fused QKV projection.  Y[m,n] = sum_k X[m,k]*W[n,k] + bias[n]
// Output written f16 in (B,H,S,DK) layout.  blockIdx.z selects Q/K/V.
// ---------------------------------------------------------------------------
__global__ void qkv_proj_kernel(
    const float* __restrict__ q_in, const float* __restrict__ k_in,
    const float* __restrict__ v_in,
    const float* __restrict__ wq, const float* __restrict__ bq,
    const float* __restrict__ wk, const float* __restrict__ bk,
    const float* __restrict__ wv, const float* __restrict__ bv,
    _Float16* __restrict__ q_out, _Float16* __restrict__ k_out,
    _Float16* __restrict__ v_out) {
  const int lane = threadIdx.x & 31;
  const int wave = threadIdx.x >> 5;
  const int which = blockIdx.z;

  const float* x    = which == 0 ? q_in : (which == 1 ? k_in : v_in);
  const float* w    = which == 0 ? wq   : (which == 1 ? wk   : wv);
  const float* bias = which == 0 ? bq   : (which == 1 ? bk   : bv);
  _Float16* out     = which == 0 ? q_out : (which == 1 ? k_out : v_out);

  const int m0 = blockIdx.y * 16;
  const int n0 = (blockIdx.x * 4 + wave) * 16;

  v8f acc = {};
  for (int k0 = 0; k0 < D_MODEL; k0 += 32) {
    v16h a = frag_A_f32(x, D_MODEL, m0, k0, lane);
    v16h b = frag_B_nk_f32(w, D_MODEL, n0, k0, lane);
    acc = __builtin_amdgcn_wmma_f32_16x16x32_f16(false, a, false, b,
                                                 (short)0, acc, false, false);
  }

  const int lm = lane & 15, lhi = lane >> 4;
  const int col = n0 + lm;
  const int h = col >> 6, d = col & 63;
  const float bcol = bias[col];
#pragma unroll
  for (int r = 0; r < 8; ++r) {
    const int m = m0 + r + lhi * 8;
    const int bb = m >> 11;              // batch
    const int s = m & (SEQ - 1);         // seq position
    out[((size_t)(bb * NH + h) * SEQ + s) * DK + d] = (_Float16)(acc[r] + bcol);
  }
}

// ---------------------------------------------------------------------------
// Kernel 2: fused attention per (b,h, 32-query strip).
// 256 KB LDS score strip + 4 KB TDM-staged V block (CDNA5 320 KB WGP LDS).
// scores (WMMA) -> masked scale -> wave32 softmax -> attn write (once) ->
// ctx = P @ V (WMMA over TDM-staged V) -> ctx f16 in (B,S,D_MODEL) layout.
// ---------------------------------------------------------------------------
__global__ void attention_kernel(
    const _Float16* __restrict__ Q, const _Float16* __restrict__ K,
    const _Float16* __restrict__ V, const int* __restrict__ mask,
    float* __restrict__ attn, _Float16* __restrict__ ctx) {
  extern __shared__ float sc[];            // 32 x 2048 f32 strip, then V stage

  const int lane = threadIdx.x & 31;
  const int wave = threadIdx.x >> 5;       // 4 waves / block
  const int bh = blockIdx.y;               // 0..31  (b*16 + h)
  const int b = bh >> 4;
  const int hh = bh & 15;
  const int q0 = blockIdx.x * 32;

  const _Float16* Qb = Q + (size_t)bh * SEQ * DK;
  const _Float16* Kb = K + (size_t)bh * SEQ * DK;
  const _Float16* Vb = V + (size_t)bh * SEQ * DK;
  const int* mb = mask + b * SEQ;
  const int lm = lane & 15, lhi = lane >> 4;

  // ---- Stage A: scores = scale * Q K^T, with mask, into LDS strip --------
  for (int t = wave; t < 256; t += 4) {    // 2 q-subtiles x 128 k-tiles
    const int qi = t & 1;
    const int n0 = (t >> 1) * 16;
    const int m0 = q0 + qi * 16;
    v8f acc = {};
#pragma unroll
    for (int kk = 0; kk < DK; kk += 32) {
      v16h a = frag_A_f16(Qb, DK, m0, kk, lane);
      v16h bt = frag_B_nk_f16(Kb, DK, n0, kk, lane);   // B(k,n) = K[n,k]
      acc = __builtin_amdgcn_wmma_f32_16x16x32_f16(false, a, false, bt,
                                                   (short)0, acc, false, false);
    }
    const int col = n0 + lm;
    const bool mz = (mb[col] == 0);
#pragma unroll
    for (int r = 0; r < 8; ++r) {
      const int lr = qi * 16 + r + lhi * 8;            // local row 0..31
      const float sv = acc[r] * 0.125f;                // 1/sqrt(64)
      sc[(size_t)lr * SEQ + col] = mz ? NEG_INF : sv;
    }
  }
  __syncthreads();

  // ---- Stage B: row softmax (wave32 shuffle reductions), write attn ------
  for (int r = wave * 8; r < wave * 8 + 8; ++r) {
    float* rowp = sc + (size_t)r * SEQ;
    float mx = -3.4e38f;
    for (int c = lane; c < SEQ; c += 32) mx = fmaxf(mx, rowp[c]);
#pragma unroll
    for (int off = 16; off > 0; off >>= 1) mx = fmaxf(mx, __shfl_xor(mx, off, 32));
    float sum = 0.0f;
    for (int c = lane; c < SEQ; c += 32) {
      const float e = __expf(rowp[c] - mx);
      rowp[c] = e;
      sum += e;
    }
#pragma unroll
    for (int off = 16; off > 0; off >>= 1) sum += __shfl_xor(sum, off, 32);
    const float inv = 1.0f / sum;
    float* arow = attn + ((size_t)bh * SEQ + (q0 + r)) * SEQ;
    for (int c = lane; c < SEQ; c += 32) {
      const float p = rowp[c] * inv;
      rowp[c] = p;                 // keep normalized P in LDS for ctx stage
      arow[c] = p;                 // coalesced single write of attn output
    }
  }

  // ---- Stage C: ctx = P(32x2048) @ V(2048x64), V staged via TDM ----------
  // Wave tile assignment: qi = wave&1; two n-tiles n00, n00+32 per wave.
  _Float16* vbuf = (_Float16*)(sc + 32 * SEQ);   // 32 x 64 f16 = 4 KB
  const int qi  = wave & 1;
  const int n00 = (wave >> 1) * 16;
  const int n01 = n00 + 32;
  v8f acc0 = {}, acc1 = {};

  for (int kk = 0; kk < SEQ; kk += 32) {
    __syncthreads();                       // vbuf from prev iter fully read
    if (wave == 0) {
      // DMA V[kk..kk+32) x 64 (f16, row stride 64) into LDS via TDM.
      tdm_load_2d_f16((unsigned)(uintptr_t)vbuf, Vb + (size_t)kk * DK,
                      32, DK, DK);
      __builtin_amdgcn_s_wait_tensorcnt(0);
    }
    __syncthreads();                       // vbuf ready for all waves

    v16h a  = frag_A_lds(sc, SEQ, qi * 16, kk, lane);      // P (f32->f16)
    v16h b0 = frag_B_kn_lds(vbuf, DK, 0, n00, lane);       // B(k,n)=V[k,n]
    v16h b1 = frag_B_kn_lds(vbuf, DK, 0, n01, lane);
    acc0 = __builtin_amdgcn_wmma_f32_16x16x32_f16(false, a, false, b0,
                                                  (short)0, acc0, false, false);
    acc1 = __builtin_amdgcn_wmma_f32_16x16x32_f16(false, a, false, b1,
                                                  (short)0, acc1, false, false);
  }

  const int cbase = hh * DK;
#pragma unroll
  for (int r = 0; r < 8; ++r) {
    const int srow = q0 + qi * 16 + r + lhi * 8;
    const size_t grow = (size_t)b * SEQ + srow;
    ctx[grow * D_MODEL + cbase + n00 + lm] = (_Float16)acc0[r];
    ctx[grow * D_MODEL + cbase + n01 + lm] = (_Float16)acc1[r];
  }
}

// ---------------------------------------------------------------------------
// Kernel 3: output projection.  out[m,n] = sum_k ctx[m,k]*Wo[n,k] + bo[n]
// ---------------------------------------------------------------------------
__global__ void out_proj_kernel(const _Float16* __restrict__ ctx,
                                const float* __restrict__ wo,
                                const float* __restrict__ bo,
                                float* __restrict__ out) {
  const int lane = threadIdx.x & 31;
  const int wave = threadIdx.x >> 5;
  const int m0 = blockIdx.y * 16;
  const int n0 = (blockIdx.x * 4 + wave) * 16;

  v8f acc = {};
  for (int k0 = 0; k0 < D_MODEL; k0 += 32) {
    v16h a = frag_A_f16(ctx, D_MODEL, m0, k0, lane);
    v16h b = frag_B_nk_f32(wo, D_MODEL, n0, k0, lane);
    acc = __builtin_amdgcn_wmma_f32_16x16x32_f16(false, a, false, b,
                                                 (short)0, acc, false, false);
  }
  const int lm = lane & 15, lhi = lane >> 4;
  const int col = n0 + lm;
  const float bcol = bo[col];
#pragma unroll
  for (int r = 0; r < 8; ++r) {
    const int m = m0 + r + lhi * 8;
    out[(size_t)m * D_MODEL + col] = acc[r] + bcol;
  }
}

// ---------------------------------------------------------------------------
extern "C" void kernel_launch(void* const* d_in, const int* in_sizes, int n_in,
                              void* d_out, int out_size, void* d_ws,
                              size_t ws_size, hipStream_t stream) {
  const float* query = (const float*)d_in[0];
  const float* key   = (const float*)d_in[1];
  const float* value = (const float*)d_in[2];
  const int*   mask  = (const int*)d_in[3];
  const float* w_q = (const float*)d_in[4];
  const float* b_q = (const float*)d_in[5];
  const float* w_k = (const float*)d_in[6];
  const float* b_k = (const float*)d_in[7];
  const float* w_v = (const float*)d_in[8];
  const float* b_v = (const float*)d_in[9];
  const float* w_o = (const float*)d_in[10];
  const float* b_o = (const float*)d_in[11];

  float* out  = (float*)d_out;
  float* attn = out + (size_t)M_TOTAL * D_MODEL;   // after (2,2048,1024)

  const size_t per_tensor = (size_t)BATCH * NH * SEQ * DK;  // 4.19M f16 elems
  _Float16* q_ws   = (_Float16*)d_ws;
  _Float16* k_ws   = q_ws + per_tensor;
  _Float16* v_ws   = k_ws + per_tensor;
  _Float16* ctx_ws = v_ws + per_tensor;            // (B*S, D_MODEL) f16

  // 1) QKV projections: grid (n-tiles/4, m-tiles, 3), 4 waves/block
  qkv_proj_kernel<<<dim3(D_MODEL / 64, M_TOTAL / 16, 3), 128, 0, stream>>>(
      query, key, value, w_q, b_q, w_k, b_k, w_v, b_v, q_ws, k_ws, v_ws);

  // 2) Fused attention: 260 KB dynamic LDS (CDNA5 320 KB WGP LDS)
  const size_t lds_bytes = (size_t)32 * SEQ * sizeof(float)   // score strip
                         + (size_t)32 * DK * sizeof(_Float16); // V stage
  (void)hipFuncSetAttribute((const void*)attention_kernel,
                            hipFuncAttributeMaxDynamicSharedMemorySize,
                            (int)lds_bytes);
  attention_kernel<<<dim3(SEQ / 32, BATCH * NH), 128, lds_bytes, stream>>>(
      q_ws, k_ws, v_ws, mask, attn, ctx_ws);

  // 3) Output projection
  out_proj_kernel<<<dim3(D_MODEL / 64, M_TOTAL / 16), 128, 0, stream>>>(
      ctx_ws, w_o, b_o, out);
}
